// MultiModalTransformerSharedEncoder_22376779612331
// MI455X (gfx1250) — compile-verified, matrix operand order checked by reading, not run on hardware
//
#include <hip/hip_runtime.h>
#include <hip/hip_bf16.h>
#include <math.h>
#include <stdint.h>

// ---------------------------------------------------------------------------
// MultiModalTransformerSharedEncoder for MI455X (gfx1250, wave32, WMMA).
// GEMM-shaped compute uses v_wmma_f32_16x16x32_f16 (f16 in, f32 acc).
// Weights pre-transposed/converted to f16 once; activations keep an f16
// mirror so all WMMA staging is pure-copy -> staged with CDNA5 async
// global->LDS copies (ASYNCcnt) and double-buffered LDS tiles.
// ---------------------------------------------------------------------------

typedef __attribute__((ext_vector_type(16))) _Float16 v16h;
typedef __attribute__((ext_vector_type(8)))  _Float16 v8h;
typedef __attribute__((ext_vector_type(4)))  _Float16 v4h;
typedef __attribute__((ext_vector_type(8)))  float    v8f;

#define FEAT   256
#define NHEAD  8
#define HD     32
#define NL     3
#define Bsz    16
#define SLEN   500
#define TP     512            // padded T (501 -> 512)
#define NSEQ   32             // 2*B
#define MROWS  (NSEQ * TP)    // 16384
#define FF     1024

// LDS tile row stride in halves: 40 halves = 80 bytes -> conflict-free 16B reads
#define TS 40

// ---- CDNA5 async global->LDS copy (tracked by ASYNCcnt) -------------------
__device__ __forceinline__ void async_copy_b128(const void* gptr, void* lptr) {
  unsigned lds = (unsigned)(uintptr_t)lptr;              // low 32b = LDS offset
  unsigned long long ga = (unsigned long long)(uintptr_t)gptr;
  asm volatile("global_load_async_to_lds_b128 %0, %1, off"
               :: "v"(lds), "v"(ga) : "memory");
}
__device__ __forceinline__ void async_wait0() {
  asm volatile("s_wait_asynccnt 0x0" ::: "memory");
}

// Build a 16-half WMMA A/B fragment from LDS.
// Element e holds k = 16*(e>>3) + kh + (e&7); lo chunk at base+kh, hi at base+16+kh.
__device__ __forceinline__ v16h load_frag(const _Float16* rowbase, int kh) {
  v8h lo = *(const v8h*)(rowbase + kh);
  v8h hi = *(const v8h*)(rowbase + 16 + kh);
  v16h f;
#pragma unroll
  for (int e = 0; e < 8; ++e) { f[e] = lo[e]; f[8 + e] = hi[e]; }
  return f;
}

// ---------------------------------------------------------------------------
// Weight prep: in[K][N] f32 -> out[N][K] f16 (transposed, write-coalesced).
// ---------------------------------------------------------------------------
__global__ __launch_bounds__(256)
void wcvt_t(const float* __restrict__ in, _Float16* __restrict__ out, int K, int N) {
  int idx = blockIdx.x * 256 + threadIdx.x;
  if (idx >= K * N) return;
  int n = idx / K, k = idx - n * K;
  out[idx] = (_Float16)in[(size_t)k * N + n];
}

// ---------------------------------------------------------------------------
// GEMM: C[M,N] = act(A[M,K] @ W[K,N] + bias), W pre-transposed f16 Wt[N][K].
// TA in {_Float16,float}; TC in {_Float16,float}.  128 threads (4 waves);
// 64x64 tile/block; wave w owns rows [16w,16w+16).  act: 0=none, 1=exact GELU.
// EDGE=false fast path: M,N covered, K%32==0, f16 A rows 16B aligned, async
// staging + double-buffered LDS (1 barrier per k-step).
// ---------------------------------------------------------------------------
template <bool EDGE, typename TA, typename TC>
__global__ __launch_bounds__(128)
void gemm_wmma(const TA* __restrict__ A, int lda,
               const _Float16* __restrict__ Wt,
               const float* __restrict__ bias,
               TC* __restrict__ C, int ldc,
               int M, int N, int K, int act) {
  __shared__ __align__(16) _Float16 As[2][64 * TS];
  __shared__ __align__(16) _Float16 Ws[2][64 * TS];   // Ws[n][k]
  const int tid  = threadIdx.x;
  const int lane = tid & 31;
  const int wave = tid >> 5;
  const int l15  = lane & 15;
  const int kh   = (lane >> 4) * 8;
  const int m0   = blockIdx.x * 64;
  const int n0   = blockIdx.y * 64;

  v8f acc[4];
#pragma unroll
  for (int j = 0; j < 4; ++j) acc[j] = (v8f)(0.f);

  if constexpr (!EDGE) {
    // ---- stage tile (async, pure f16 copies, 16B granules) ----
    auto stage = [&](int b, int k0) {
#pragma unroll
      for (int i = 0; i < 2; ++i) {
        int idx = tid + i * 128;
        int r = idx >> 2, c8 = (idx & 3) * 8;
        async_copy_b128((const _Float16*)A + (size_t)(m0 + r) * lda + k0 + c8,
                        &As[b][r * TS + c8]);
        async_copy_b128(Wt + (size_t)(n0 + r) * K + k0 + c8,
                        &Ws[b][r * TS + c8]);
      }
    };
    stage(0, 0);
    async_wait0();
    __syncthreads();
    int buf = 0;
    for (int k0 = 0; k0 < K; k0 += 32, buf ^= 1) {
      if (k0 + 32 < K) stage(buf ^ 1, k0 + 32);   // overlap with WMMAs below
      v16h a = load_frag(&As[buf][(wave * 16 + l15) * TS], kh);
#pragma unroll
      for (int j = 0; j < 4; ++j) {
        v16h b = load_frag(&Ws[buf][(j * 16 + l15) * TS], kh);
        acc[j] = __builtin_amdgcn_wmma_f32_16x16x32_f16(
            false, a, false, b, (short)0, acc[j], false, false);
      }
      async_wait0();
      __syncthreads();
    }
  } else {
    // ---- guarded staging (ragged K), single buffer, force-unrolled ----
    for (int k0 = 0; k0 < K; k0 += 32) {
#pragma unroll
      for (int ii = 0; ii < 16; ++ii) {
        int i = tid + ii * 128;
        int r = i >> 5, c = i & 31;
        int gm = m0 + r, gk = k0 + c;
        float v = (gm < M && gk < K) ? (float)A[(size_t)gm * lda + gk] : 0.f;
        As[0][r * TS + c] = (_Float16)v;
      }
#pragma unroll
      for (int ii = 0; ii < 16; ++ii) {
        int i = tid + ii * 128;
        int nr = i >> 5, c = i & 31;
        int gn = n0 + nr, gk = k0 + c;
        _Float16 v = (gn < N && gk < K) ? Wt[(size_t)gn * K + gk] : (_Float16)0.f;
        Ws[0][nr * TS + c] = v;
      }
      __syncthreads();
      v16h a = load_frag(&As[0][(wave * 16 + l15) * TS], kh);
#pragma unroll
      for (int j = 0; j < 4; ++j) {
        v16h b = load_frag(&Ws[0][(j * 16 + l15) * TS], kh);
        acc[j] = __builtin_amdgcn_wmma_f32_16x16x32_f16(
            false, a, false, b, (short)0, acc[j], false, false);
      }
      __syncthreads();
    }
  }

  // epilogue: row m = r + 8*(lane>>4), col = j*16 + (lane&15)
  const int mbase = m0 + wave * 16 + 8 * (lane >> 4);
#pragma unroll
  for (int j = 0; j < 4; ++j) {
    int gn = n0 + j * 16 + l15;
    if constexpr (EDGE) { if (gn >= N) continue; }
    float bv = bias ? bias[gn] : 0.f;
#pragma unroll
    for (int r = 0; r < 8; ++r) {
      int gm = mbase + r;
      if constexpr (EDGE) { if (gm >= M) continue; }
      float v = acc[j][r] + bv;
      if (act == 1) v = 0.5f * v * (1.f + erff(v * 0.70710678118654752f));
      C[(size_t)gm * ldc + gn] = (TC)v;
    }
  }
}

// ---------------------------------------------------------------------------
// Flash attention, one block per (qtile of 64, seq n, head h).
// qkv: f16 [NSEQ, TP, 768] (q|k|v); out: f16 [NSEQ, TP, 256].
// Key validity (== reference safe_valid): even n -> t < seq_len[n/2],
// odd n -> t == 0.  Scale applied to scores post-WMMA.  K tiles staged with
// async copies; K/V double-buffered (2 barriers per k-tile).
// ---------------------------------------------------------------------------
__global__ __launch_bounds__(128)
void flash_attn(const _Float16* __restrict__ qkv,
                const int* __restrict__ seq_len,
                _Float16* __restrict__ out) {
  __shared__ __align__(16) _Float16 Qs[64 * TS];
  __shared__ __align__(16) _Float16 Ks[2][64 * TS];
  __shared__ __align__(16) _Float16 Vt[2][32 * 72];    // [d][key], stride 72
  __shared__ __align__(16) _Float16 Pb[4][16 * 72];    // per-wave P, 16x64

  const int qt = blockIdx.x;   // 0..7
  const int n  = blockIdx.y;   // 0..31
  const int h  = blockIdx.z;   // 0..7
  const int tid = threadIdx.x, lane = tid & 31, wave = tid >> 5;
  const int l15 = lane & 15;
  const int kh  = (lane >> 4) * 8;
  const int q0  = qt * 64;
  const float scale = 0.17677669529663687f;  // 1/sqrt(HD)
  const size_t base = (size_t)n * TP;

  const int limit = (n & 1) ? 1 : seq_len[n >> 1];  // keys valid iff t < limit

  // stage K (async) and V (VGPR path, transposed) for one k-tile
  auto stageKV = [&](int b, int kt) {
#pragma unroll
    for (int i = 0; i < 2; ++i) {
      int idx = tid + i * 128;
      int r = idx >> 2, c8 = (idx & 3) * 8;
      const _Float16* rowp = qkv + (base + kt * 64 + r) * 768;
      async_copy_b128(rowp + 256 + h * HD + c8, &Ks[b][r * TS + c8]);
      v8h vv = *(const v8h*)(rowp + 512 + h * HD + c8);
#pragma unroll
      for (int e = 0; e < 8; ++e)
        Vt[b][(c8 + e) * 72 + r] = vv[e];   // transpose V -> [d][key]
    }
  };

  // stage Q tile (async pure copy)
#pragma unroll
  for (int i = 0; i < 2; ++i) {
    int idx = tid + i * 128;
    int r = idx >> 2, c8 = (idx & 3) * 8;
    async_copy_b128(qkv + (base + q0 + r) * 768 + h * HD + c8,
                    &Qs[r * TS + c8]);
  }
  stageKV(0, 0);
  async_wait0();
  __syncthreads();

  float mI[8], lI[8];
#pragma unroll
  for (int r = 0; r < 8; ++r) { mI[r] = -INFINITY; lI[r] = 0.f; }
  v8f O0 = (v8f)(0.f), O1 = (v8f)(0.f);

  int buf = 0;
  for (int kt = 0; kt < 8; ++kt, buf ^= 1) {
    if (kt + 1 < 8) stageKV(buf ^ 1, kt + 1);   // overlap with S/softmax

    // S = Q @ K^T  (B-frag of K^T == row-major K tile in LDS)
    v16h aq = load_frag(&Qs[(wave * 16 + l15) * TS], kh);
    v8f S[4];
#pragma unroll
    for (int j = 0; j < 4; ++j) {
      v16h bk = load_frag(&Ks[buf][(j * 16 + l15) * TS], kh);
      S[j] = __builtin_amdgcn_wmma_f32_16x16x32_f16(
          false, aq, false, bk, (short)0, (v8f)(0.f), false, false);
    }

    // scale + mask + per-row max (row r+8*(lane>>4) lives in 16-lane group)
    float rmax[8];
#pragma unroll
    for (int r = 0; r < 8; ++r) rmax[r] = -INFINITY;
#pragma unroll
    for (int j = 0; j < 4; ++j) {
      int tkey = kt * 64 + j * 16 + l15;
      bool ok = tkey < limit;
#pragma unroll
      for (int r = 0; r < 8; ++r) {
        float s = ok ? S[j][r] * scale : -INFINITY;
        S[j][r] = s;
        rmax[r] = fmaxf(rmax[r], s);
      }
    }
#pragma unroll
    for (int off = 1; off < 16; off <<= 1)
#pragma unroll
      for (int r = 0; r < 8; ++r)
        rmax[r] = fmaxf(rmax[r], __shfl_xor(rmax[r], off, 32));

    float alpha[8];
#pragma unroll
    for (int r = 0; r < 8; ++r) {
      float mNew = fmaxf(mI[r], rmax[r]);
      alpha[r] = __expf(mI[r] - mNew);   // exp(-inf)=0 on first tile
      mI[r] = mNew;
    }

    // P = exp(S - m); stash into LDS in A-frag friendly layout
    float rsum[8];
#pragma unroll
    for (int r = 0; r < 8; ++r) rsum[r] = 0.f;
#pragma unroll
    for (int j = 0; j < 4; ++j)
#pragma unroll
      for (int r = 0; r < 8; ++r) {
        float p = __expf(S[j][r] - mI[r]);   // masked -> exp(-inf)=0
        rsum[r] += p;
        Pb[wave][(r + 8 * (lane >> 4)) * 72 + j * 16 + l15] = (_Float16)p;
      }
#pragma unroll
    for (int off = 1; off < 16; off <<= 1)
#pragma unroll
      for (int r = 0; r < 8; ++r)
        rsum[r] += __shfl_xor(rsum[r], off, 32);
#pragma unroll
    for (int r = 0; r < 8; ++r) {
      lI[r] = lI[r] * alpha[r] + rsum[r];
      O0[r] *= alpha[r];
      O1[r] *= alpha[r];
    }
    async_wait0();
    __syncthreads();   // publish P + next K tile

    // O += P @ V   (K dim = 64 keys -> 2 WMMA k-steps)
#pragma unroll
    for (int ks = 0; ks < 2; ++ks) {
      v16h ap = load_frag(&Pb[wave][l15 * 72 + ks * 32], kh);
      v16h b0 = load_frag(&Vt[buf][(0 * 16 + l15) * 72 + ks * 32], kh);
      v16h b1 = load_frag(&Vt[buf][(1 * 16 + l15) * 72 + ks * 32], kh);
      O0 = __builtin_amdgcn_wmma_f32_16x16x32_f16(false, ap, false, b0,
                                                  (short)0, O0, false, false);
      O1 = __builtin_amdgcn_wmma_f32_16x16x32_f16(false, ap, false, b1,
                                                  (short)0, O1, false, false);
    }
    __syncthreads();   // P@V reads done before next overwrite of buf
  }

  // epilogue: divide by row sums and store f16 [n][tok][h*32+d]
  const int mloc = wave * 16 + 8 * (lane >> 4);
#pragma unroll
  for (int r = 0; r < 8; ++r) {
    int tok = q0 + mloc + r;
    float inv = 1.f / lI[r];
    out[(base + tok) * FEAT + h * HD + 0 * 16 + l15] = (_Float16)(O0[r] * inv);
    out[(base + tok) * FEAT + h * HD + 1 * 16 + l15] = (_Float16)(O1[r] * inv);
  }
}

// ---------------------------------------------------------------------------
// Small VALU kernels
// ---------------------------------------------------------------------------
__global__ __launch_bounds__(256)
void zero_kernel(float* p, int nElem) {
  int i = blockIdx.x * 256 + threadIdx.x;
  if (i < nElem) p[i] = 0.f;
}

// x[2b][s][:] = proj[b*500+s][:] + ehr_b + modality_emb[0] + pos_enc[s]; f32+f16
__global__ __launch_bounds__(256)
void scatter_embed(const float* __restrict__ proj, const float* __restrict__ eb,
                   const float* __restrict__ me, const float* __restrict__ pos,
                   float* __restrict__ x, _Float16* __restrict__ x16) {
  int i = blockIdx.x;          // 0..7999
  int d = threadIdx.x;
  int b = i / SLEN, s = i - b * SLEN;
  float v = proj[(size_t)i * FEAT + d] + eb[d] + me[d] + pos[(size_t)s * FEAT + d];
  size_t o = (((size_t)(2 * b) * TP) + s) * FEAT + d;
  x[o] = v;
  x16[o] = (_Float16)v;
}

// x[2b+1][0][:] = cxr[b] @ cxr_w + cxr_b + modality_emb[1] + pos_enc[0]
__global__ __launch_bounds__(256)
void cxr_embed(const float* __restrict__ cxr, const float* __restrict__ w,
               const float* __restrict__ bias, const float* __restrict__ me,
               const float* __restrict__ pos,
               float* __restrict__ x, _Float16* __restrict__ x16) {
  int b = blockIdx.x, d = threadIdx.x;
  float s = bias[d] + me[FEAT + d] + pos[d];
  for (int k = 0; k < 2048; ++k)
    s += cxr[(size_t)b * 2048 + k] * w[(size_t)k * FEAT + d];
  size_t o = ((size_t)(2 * b + 1) * TP) * FEAT + d;
  x[o] = s;
  x16[o] = (_Float16)s;
}

// x_tok = LayerNorm(x_tok + y_tok) * g + b   (writes f32 + f16 mirror)
__global__ __launch_bounds__(256)
void residual_ln(float* __restrict__ x, const float* __restrict__ y,
                 const float* __restrict__ g, const float* __restrict__ b,
                 _Float16* __restrict__ x16) {
  __shared__ float red[256];
  size_t t = blockIdx.x;
  int d = threadIdx.x;
  float v = x[t * FEAT + d] + y[t * FEAT + d];
  red[d] = v;
  __syncthreads();
  for (int s = 128; s > 0; s >>= 1) {
    if (d < s) red[d] += red[d + s];
    __syncthreads();
  }
  float mu = red[0] * (1.f / FEAT);
  __syncthreads();
  float c = v - mu;
  red[d] = c * c;
  __syncthreads();
  for (int s = 128; s > 0; s >>= 1) {
    if (d < s) red[d] += red[d + s];
    __syncthreads();
  }
  float var = red[0] * (1.f / FEAT);
  float r = c * rsqrtf(var + 1e-5f) * g[d] + b[d];
  x[t * FEAT + d] = r;
  x16[t * FEAT + d] = (_Float16)r;
}

// masked mean pool over valid tokens
__global__ __launch_bounds__(256)
void pool_kernel(const float* __restrict__ x, const int* __restrict__ seq_len,
                 const int* __restrict__ valid_cxr, float* __restrict__ pooled) {
  int n = blockIdx.x, d = threadIdx.x;
  int cnt = (n & 1) ? (valid_cxr[n >> 1] ? 1 : 0) : seq_len[n >> 1];
  float s = 0.f;
  for (int t = 0; t < cnt; ++t) s += x[((size_t)n * TP + t) * FEAT + d];
  pooled[n * FEAT + d] = (cnt > 0) ? s / (float)cnt : 0.f;
}

// out = pooled @ outp_w + outp_b, duplicated (shared_ft, shared_gft)
__global__ __launch_bounds__(256)
void out_proj(const float* __restrict__ pooled, const float* __restrict__ w,
              const float* __restrict__ b, float* __restrict__ out) {
  int n = blockIdx.x, d = threadIdx.x;
  float s = b[d];
  for (int k = 0; k < FEAT; ++k)
    s += pooled[n * FEAT + k] * w[(size_t)k * FEAT + d];
  out[n * FEAT + d] = s;
  out[NSEQ * FEAT + n * FEAT + d] = s;
}

// ---------------------------------------------------------------------------
extern "C" void kernel_launch(void* const* d_in, const int* in_sizes, int n_in,
                              void* d_out, int out_size, void* d_ws, size_t ws_size,
                              hipStream_t stream) {
  const float* ehr_data   = (const float*)d_in[0];   // [16,500,498]
  const float* cxr_data   = (const float*)d_in[1];   // [16,2048]
  const int*   seq_len    = (const int*)d_in[2];     // [16]
  const int*   valid_cxr  = (const int*)d_in[3];     // [16]
  const float* ehr_w      = (const float*)d_in[4];   // [498,256]
  const float* ehr_b      = (const float*)d_in[5];
  const float* cxr_w      = (const float*)d_in[6];   // [2048,256]
  const float* cxr_b      = (const float*)d_in[7];
  const float* mod_emb    = (const float*)d_in[8];   // [2,256]
  const float* pos_enc    = (const float*)d_in[9];   // [1,500,256]
  const float* qkv_w      = (const float*)d_in[10];  // [3,256,768]
  const float* qkv_b      = (const float*)d_in[11];  // [3,768]
  const float* ao_w       = (const float*)d_in[12];  // [3,256,256]
  const float* ao_b       = (const float*)d_in[13];
  const float* ln1_g      = (const float*)d_in[14];
  const float* ln1_b      = (const float*)d_in[15];
  const float* ln2_g      = (const float*)d_in[16];
  const float* ln2_b      = (const float*)d_in[17];
  const float* ff1_w      = (const float*)d_in[18];  // [3,256,1024]
  const float* ff1_b      = (const float*)d_in[19];
  const float* ff2_w      = (const float*)d_in[20];  // [3,1024,256]
  const float* ff2_b      = (const float*)d_in[21];
  const float* outp_w     = (const float*)d_in[22];  // [256,256]
  const float* outp_b     = (const float*)d_in[23];

  // ---- workspace carve-up (all chunk sizes are multiples of 16B) ----
  char* w = (char*)d_ws;
  float*     x      = (float*)w;      w += (size_t)MROWS * FEAT * 4;   // f32 residual
  _Float16*  x16    = (_Float16*)w;   w += (size_t)MROWS * FEAT * 2;   // f16 mirror
  _Float16*  qkv16  = (_Float16*)w;   w += (size_t)MROWS * 3 * FEAT * 2;
  _Float16*  ao16   = (_Float16*)w;   w += (size_t)MROWS * FEAT * 2;   // attn out (f16)
  float*     yf     = (float*)w;      w += (size_t)MROWS * FEAT * 4;   // f32 gemm out
  _Float16*  h16    = (_Float16*)w;   w += (size_t)MROWS * FF * 2;     // ff1 out (f16)
  _Float16*  ehrT   = (_Float16*)w;   w += (size_t)FEAT * 498 * 2;
  _Float16*  qkvT   = (_Float16*)w;   w += (size_t)NL * 3 * FEAT * FEAT * 2;
  _Float16*  aoT    = (_Float16*)w;   w += (size_t)NL * FEAT * FEAT * 2;
  _Float16*  ff1T   = (_Float16*)w;   w += (size_t)NL * FF * FEAT * 2;
  _Float16*  ff2T   = (_Float16*)w;   w += (size_t)NL * FEAT * FF * 2;
  float*     pooled = (float*)w;

  // ---- weight prep: transpose + f16 convert (once per launch) ----
  wcvt_t<<<(498 * FEAT + 255) / 256, 256, 0, stream>>>(ehr_w, ehrT, 498, FEAT);
  for (int l = 0; l < NL; ++l) {
    wcvt_t<<<(FEAT * 3 * FEAT + 255) / 256, 256, 0, stream>>>(
        qkv_w + (size_t)l * FEAT * 3 * FEAT, qkvT + (size_t)l * 3 * FEAT * FEAT, FEAT, 3 * FEAT);
    wcvt_t<<<(FEAT * FEAT + 255) / 256, 256, 0, stream>>>(
        ao_w + (size_t)l * FEAT * FEAT, aoT + (size_t)l * FEAT * FEAT, FEAT, FEAT);
    wcvt_t<<<(FEAT * FF + 255) / 256, 256, 0, stream>>>(
        ff1_w + (size_t)l * FEAT * FF, ff1T + (size_t)l * FF * FEAT, FEAT, FF);
    wcvt_t<<<(FF * FEAT + 255) / 256, 256, 0, stream>>>(
        ff2_w + (size_t)l * FF * FEAT, ff2T + (size_t)l * FEAT * FF, FF, FEAT);
  }

  // zero residual stream + f16 mirror (pad tokens / empty slots must be 0)
  zero_kernel<<<(MROWS * FEAT + 255) / 256, 256, 0, stream>>>(x, MROWS * FEAT);
  zero_kernel<<<(MROWS * FEAT / 2 + 255) / 256, 256, 0, stream>>>((float*)x16, MROWS * FEAT / 2);

  // ehr projection: [8000,498] @ [498,256] -> yf (ragged K -> EDGE path)
  {
    dim3 g((Bsz * SLEN + 63) / 64, (FEAT + 63) / 64);
    gemm_wmma<true, float, float><<<g, 128, 0, stream>>>(
        ehr_data, 498, ehrT, nullptr, yf, FEAT, Bsz * SLEN, FEAT, 498, 0);
  }
  scatter_embed<<<Bsz * SLEN, 256, 0, stream>>>(yf, ehr_b, mod_emb, pos_enc, x, x16);
  cxr_embed<<<Bsz, 256, 0, stream>>>(cxr_data, cxr_w, cxr_b, mod_emb, pos_enc, x, x16);

  for (int l = 0; l < NL; ++l) {
    // QKV: x16 [16384,256] @ [256,768] -> qkv16 (f16)
    {
      dim3 g(MROWS / 64, (3 * FEAT) / 64);
      gemm_wmma<false, _Float16, _Float16><<<g, 128, 0, stream>>>(
          x16, FEAT, qkvT + (size_t)l * 3 * FEAT * FEAT,
          qkv_b + (size_t)l * 3 * FEAT, qkv16, 3 * FEAT,
          MROWS, 3 * FEAT, FEAT, 0);
    }
    // attention -> ao16 (f16)
    {
      dim3 g(TP / 64, NSEQ, NHEAD);
      flash_attn<<<g, 128, 0, stream>>>(qkv16, seq_len, ao16);
    }
    // attn out projection: ao16 @ [256,256] -> yf (f32)
    {
      dim3 g(MROWS / 64, FEAT / 64);
      gemm_wmma<false, _Float16, float><<<g, 128, 0, stream>>>(
          ao16, FEAT, aoT + (size_t)l * FEAT * FEAT,
          ao_b + (size_t)l * FEAT, yf, FEAT, MROWS, FEAT, FEAT, 0);
    }
    residual_ln<<<MROWS, 256, 0, stream>>>(x, yf,
                                           ln1_g + (size_t)l * FEAT,
                                           ln1_b + (size_t)l * FEAT, x16);
    // FF1 + GELU: x16 @ [256,1024] -> h16 (f16)
    {
      dim3 g(MROWS / 64, FF / 64);
      gemm_wmma<false, _Float16, _Float16><<<g, 128, 0, stream>>>(
          x16, FEAT, ff1T + (size_t)l * FF * FEAT,
          ff1_b + (size_t)l * FF, h16, FF, MROWS, FF, FEAT, 1);
    }
    // FF2: h16 @ [1024,256] -> yf (f32)
    {
      dim3 g(MROWS / 64, FEAT / 64);
      gemm_wmma<false, _Float16, float><<<g, 128, 0, stream>>>(
          h16, FF, ff2T + (size_t)l * FEAT * FF,
          ff2_b + (size_t)l * FEAT, yf, FEAT, MROWS, FEAT, FF, 0);
    }
    residual_ln<<<MROWS, 256, 0, stream>>>(x, yf,
                                           ln2_g + (size_t)l * FEAT,
                                           ln2_b + (size_t)l * FEAT, x16);
  }

  pool_kernel<<<NSEQ, 256, 0, stream>>>(x, seq_len, valid_cxr, pooled);
  out_proj<<<NSEQ, 256, 0, stream>>>(pooled, outp_w, outp_b, (float*)d_out);
}
